// MOE_24653112279328
// MI455X (gfx1250) — compile-verified, matrix operand order checked by reading, not run on hardware
//
#include <hip/hip_runtime.h>
#include <hip/hip_bf16.h>
#include <cstddef>
#include <cstdint>

// ---------------------------------------------------------------------------
// MoE (top-2, capacity-dropped) for MI455X / gfx1250, wave32.
// Heavy GEMMs: V_WMMA_F32_16X16X32_F16, 4 accumulators/wave (A-frag reuse),
// A-tiles staged via Tensor Data Mover (tensor_load_to_lds + s_wait_tensorcnt),
// B-tiles staged fp32->f16 by the workgroup.
// ---------------------------------------------------------------------------

typedef __attribute__((ext_vector_type(16))) _Float16 v16h;
typedef __attribute__((ext_vector_type(8)))  _Float16 v8h;
typedef __attribute__((ext_vector_type(4)))  _Float16 v4h;
typedef __attribute__((ext_vector_type(8)))  float    v8f;

#define NB   8192      // tokens = B*T
#define CD   1024      // C
#define NE   8         // experts
#define HD   2730      // H
#define HP   2816      // H padded to multiple of 128 (N-tile)
#define CAP  2048      // capacity = ceil(2*N/E)
#define NA   (NB*2)    // assignments = N*K

// workspace byte offsets
#define WS_AUX    ((size_t)0)                       // imp[8]@0, cnt[8]@32, dropped(int)@64
#define WS_TOPKI  ((size_t)256)
#define WS_TOPKP  (WS_TOPKI + (size_t)NA*4)
#define WS_TOK    (WS_TOPKP + (size_t)NA*4)
#define WS_P      (WS_TOK   + (size_t)NE*CAP*4)
#define WS_BUFH   (WS_P     + (size_t)NE*CAP*4)
#define WS_HBUF   (WS_BUFH  + (size_t)NE*CAP*CD*2)

// output float offsets: (y, load_loss, load_frac[8], importance[8], n_dropped)
#define OUT_LOSS  (NB*CD)
#define OUT_FRAC  (OUT_LOSS+1)
#define OUT_IMP   (OUT_FRAC+8)
#define OUT_DROP  (OUT_IMP+8)

// ---------------------------------------------------------------------------
// TDM availability (5-arg builtin on ROCm7.2/clang-22, 6-arg on clang-23+).
#if defined(__has_builtin)
#  if __has_builtin(__builtin_amdgcn_tensor_load_to_lds) && \
      __has_builtin(__builtin_amdgcn_s_wait_tensorcnt)
#    define HAVE_TDM 1
#  else
#    define HAVE_TDM 0
#  endif
#else
#  define HAVE_TDM 0
#endif

#if HAVE_TDM
typedef __attribute__((ext_vector_type(4))) unsigned int u32x4;
typedef __attribute__((ext_vector_type(4))) int          i32x4;
typedef __attribute__((ext_vector_type(8))) int          i32x8;

// Issue one TDM load of a 64x64 f16 tile (row stride = stride_elems) into LDS
// at byte offset lds_off, with LDS padding of 4 DWORDs (8 halves) every
// 32 DWORDs (64 halves) so it lands in a [64][72] padded tile.
// D# layout per CDNA5 ISA.
__device__ __forceinline__ void tdm_load_tile64x64(uint32_t lds_off,
                                                   const _Float16* gptr,
                                                   uint32_t stride_elems) {
    const unsigned long long ga = (unsigned long long)(uintptr_t)gptr;
    u32x4 g0;
    g0.x = 1u;                                    // count=1 (valid descriptor)
    g0.y = lds_off;                               // lds_addr [63:32]
    g0.z = (unsigned int)ga;                      // global_addr[31:0]
    g0.w = (unsigned int)((ga >> 32) & 0x01FFFFFFull) | (2u << 30); // [56:32]|type=2
    i32x8 g1;
    g1[0] = (1 << 16)        // data_size = 1 -> 2 bytes
          | (1 << 20)        // pad_enable
          | (4 << 22)        // pad_interval: 2^(4+1)=32 DWORDs (=64 halves)
          | (3 << 25);       // pad_amount: 4 DWORDs (=8 halves)
    g1[1] = (64 << 16);      // tensor_dim0 = 64  (bits 79:48)
    g1[2] = (64 << 16);      // tensor_dim1 = 64  (bits 111:80)
    g1[3] = (64 << 16);      // tile_dim0   = 64  (bits 127:112)
    g1[4] = 64;              // tile_dim1=64, tile_dim2=0
    g1[5] = (int)stride_elems; // tensor_dim0_stride[31:0]
    g1[6] = 0;               // stride0 hi / stride1 lo
    g1[7] = 0;
    const i32x4 z4 = {0, 0, 0, 0};
#if __clang_major__ >= 23
    const i32x8 z8 = {0, 0, 0, 0, 0, 0, 0, 0};
    __builtin_amdgcn_tensor_load_to_lds(g0, g1, z4, z4, z8, 0);
#else
    __builtin_amdgcn_tensor_load_to_lds(g0, g1, z4, z4, 0);
#endif
}
#endif // HAVE_TDM

// ---------------------------------------------------------------------------
__global__ void k_init(float* __restrict__ y, float* __restrict__ aux) {
    size_t i = ((size_t)blockIdx.x*256 + threadIdx.x) * 4;   // 8192 blocks cover NB*CD
    float4 z = make_float4(0.f, 0.f, 0.f, 0.f);
    *(float4*)(y + i) = z;
    if (blockIdx.x == 0 && threadIdx.x < 24) aux[threadIdx.x] = 0.f;
}

// ---------------------------------------------------------------------------
// One wave32 per token: logits -> softmax -> top-2 -> normalized probs.
__global__ void k_gate(const float* __restrict__ x, const float* __restrict__ gw,
                       int* __restrict__ topki, float* __restrict__ topkp,
                       float* __restrict__ aux) {
    const int wid = threadIdx.x >> 5, lane = threadIdx.x & 31;
    const int t = blockIdx.x * 8 + wid;
    const float* xr = x + (size_t)t * CD;

    float acc[NE];
#pragma unroll
    for (int e = 0; e < NE; e++) acc[e] = 0.f;
    for (int c = lane; c < CD; c += 32) {
        const float xv = xr[c];
#pragma unroll
        for (int e = 0; e < NE; e++) acc[e] += xv * gw[e*CD + c];
    }
#pragma unroll
    for (int e = 0; e < NE; e++)
        for (int off = 16; off > 0; off >>= 1)
            acc[e] += __shfl_xor(acc[e], off, 32);

    if (lane == 0) {
        float mx = acc[0];
        for (int e = 1; e < NE; e++) mx = fmaxf(mx, acc[e]);
        float p[NE], s = 0.f;
        for (int e = 0; e < NE; e++) { p[e] = __expf(acc[e]-mx); s += p[e]; }
        const float inv = 1.f / s;
        for (int e = 0; e < NE; e++) p[e] *= inv;

        int i0 = 0; float p0 = p[0];
        for (int e = 1; e < NE; e++) if (p[e] > p0) { p0 = p[e]; i0 = e; }
        int i1 = -1; float p1 = -1.f;
        for (int e = 0; e < NE; e++) if (e != i0 && p[e] > p1) { p1 = p[e]; i1 = e; }

        const float d = 1.f / (p0 + p1 + 1e-9f);
        topki[t*2]   = i0;  topki[t*2+1] = i1;
        topkp[t*2]   = p0*d; topkp[t*2+1] = p1*d;

        for (int e = 0; e < NE; e++) atomicAdd(&aux[e], p[e]);  // importance sums
        atomicAdd(&aux[8 + i0], 1.f);                           // top-1 load count
    }
}

// ---------------------------------------------------------------------------
// One block per expert; deterministic prefix-scan over assignments in token
// order (matches reference stable-argsort capacity dropping).
__global__ void k_route(const int* __restrict__ topki, const float* __restrict__ topkp,
                        int* __restrict__ tok_slot, float* __restrict__ p_slot,
                        int* __restrict__ dropped) {
    const int e = blockIdx.x, tid = threadIdx.x;
    __shared__ int sc[256];

    for (int i = tid; i < CAP; i += 256) { tok_slot[e*CAP+i] = NB; p_slot[e*CAP+i] = 0.f; }
    __syncthreads();

    int running = 0;
    for (int base = 0; base < NA; base += 256) {
        const int a = base + tid;
        int match = 0, tok = 0; float pp = 0.f;
        if (topki[a] == e) { match = 1; tok = a >> 1; pp = topkp[a]; }
        sc[tid] = match;
        __syncthreads();
        for (int off = 1; off < 256; off <<= 1) {
            const int v = (tid >= off) ? sc[tid-off] : 0;
            __syncthreads();
            sc[tid] += v;
            __syncthreads();
        }
        const int pos = running + sc[tid] - match;
        if (match) {
            if (pos < CAP) { tok_slot[e*CAP+pos] = tok; p_slot[e*CAP+pos] = pp; }
            else atomicAdd(dropped, 1);
        }
        const int bt = sc[255];
        __syncthreads();
        running += bt;
    }
}

// ---------------------------------------------------------------------------
__global__ void k_gather(const float* __restrict__ x, const int* __restrict__ tok_slot,
                         _Float16* __restrict__ bufh) {
    const int row = blockIdx.x;            // e*CAP + slot
    const int tok = tok_slot[row];
    _Float16* dst = bufh + (size_t)row * CD;
    if (tok < NB) {
        const float* src = x + (size_t)tok * CD;
        for (int c = threadIdx.x; c < CD; c += 256) dst[c] = (_Float16)src[c];
    } else {
        for (int c = threadIdx.x; c < CD; c += 256) dst[c] = (_Float16)0.f;
    }
}

// ---------------------------------------------------------------------------
// WMMA 16-bit A/B fragment: lane L holds K = {8*khi+0..7, 8*khi+16..23} of
// its row/column (L&15).
__device__ __forceinline__ v16h frag16(const _Float16* rowk, int khi) {
    const v8h lo = *(const v8h*)(rowk + 8*khi);
    const v8h hi = *(const v8h*)(rowk + 8*khi + 16);
    return __builtin_shufflevector(lo, hi, 0,1,2,3,4,5,6,7,8,9,10,11,12,13,14,15);
}

// ---------------------------------------------------------------------------
// GEMM1: h = silu(buf @ w1^T + b1).  M=CAP, N=HD(->HP), K=CD.
// Block tile 64(M) x 128(N); 8 waves as 4(M) x 2(N); wave tile 16x64
// (4 accumulators, A-fragment reused 4x).
__global__ void __launch_bounds__(256) k_ffn1(
        const _Float16* __restrict__ bufh, const float* __restrict__ w1,
        const float* __restrict__ b1, _Float16* __restrict__ hbuf) {
    __shared__ __align__(16) _Float16 As[64][72];
    __shared__ __align__(16) _Float16 Bs[128][72];
    const int e     = blockIdx.z;
    const int mBase = blockIdx.y * 64;
    const int nBase = blockIdx.x * 128;
    const int tid   = threadIdx.x;
    const int w = tid >> 5, lane = tid & 31;
    const int wy = w & 3, wx = w >> 2;
    const int khi = lane >> 4, l15 = lane & 15;

    v8f cacc[4] = {};
    const _Float16* aBase = bufh + ((size_t)e*CAP + mBase) * CD;
    const float*    bRow  = w1   + (size_t)e*HD*CD;
#if HAVE_TDM
    const uint32_t asOff = (uint32_t)(uintptr_t)&As[0][0];
#endif

    for (int kb = 0; kb < CD; kb += 64) {
#if HAVE_TDM
        if (tid < 32) tdm_load_tile64x64(asOff, aBase + kb, CD);   // wave 0 drives DMA
#else
#pragma unroll
        for (int rep = 0; rep < 2; rep++) {
            const int lin = (tid + rep*256) * 8;
            const int r = lin >> 6, cc = lin & 63;
            *(v8h*)&As[r][cc] = *(const v8h*)(aBase + (size_t)r*CD + kb + cc);
        }
#endif
#pragma unroll
        for (int rep = 0; rep < 8; rep++) {   // stage B (128x64), fp32 -> f16
            const int lin = (tid + rep*256) * 4;
            const int r = lin >> 6, cc = lin & 63;
            const int nrow = nBase + r;
            v4h hv;
            if (nrow < HD) {
                const float4 f = *(const float4*)(bRow + (size_t)nrow*CD + kb + cc);
                hv.x = (_Float16)f.x; hv.y = (_Float16)f.y;
                hv.z = (_Float16)f.z; hv.w = (_Float16)f.w;
            } else {
                hv.x = hv.y = hv.z = hv.w = (_Float16)0.f;
            }
            *(v4h*)&Bs[r][cc] = hv;
        }
        __builtin_prefetch(bRow + (size_t)nBase*CD + kb + 64, 0, 0); // global_prefetch_b8
#if HAVE_TDM
        if (tid < 32) __builtin_amdgcn_s_wait_tensorcnt(0);
#endif
        __syncthreads();
#pragma unroll
        for (int kc = 0; kc < 64; kc += 32) {
            const v16h a = frag16(&As[wy*16 + l15][kc], khi);
#pragma unroll
            for (int j = 0; j < 4; j++) {
                const v16h b = frag16(&Bs[wx*64 + j*16 + l15][kc], khi);
                cacc[j] = __builtin_amdgcn_wmma_f32_16x16x32_f16(false, a, false, b,
                                                                (short)0, cacc[j],
                                                                false, false);
            }
        }
        __syncthreads();
    }

    const int mRow = mBase + wy*16 + khi*8;
#pragma unroll
    for (int j = 0; j < 4; j++) {
        const int n = nBase + wx*64 + j*16 + l15;
        const float bias = (n < HD) ? b1[e*HD + n] : 0.f;
        _Float16* hcol = hbuf + ((size_t)e*CAP + mRow) * HP + n;
#pragma unroll
        for (int i = 0; i < 8; i++) {
            const float v = cacc[j][i] + bias;
            const float s = v / (1.f + __expf(-v));         // silu
            hcol[(size_t)i*HP] = (n < HD) ? (_Float16)s : (_Float16)0.f;
        }
    }
}

// ---------------------------------------------------------------------------
// GEMM2: out = (h @ w2^T + b2) * p, scatter-add into y.  M=CAP, N=CD, K=HP.
__global__ void __launch_bounds__(256) k_ffn2(
        const _Float16* __restrict__ hbuf, const float* __restrict__ w2,
        const float* __restrict__ b2, const int* __restrict__ tok_slot,
        const float* __restrict__ p_slot, float* __restrict__ y) {
    __shared__ __align__(16) _Float16 As[64][72];
    __shared__ __align__(16) _Float16 Bs[128][72];
    const int e     = blockIdx.z;
    const int mBase = blockIdx.y * 64;
    const int nBase = blockIdx.x * 128;
    const int tid   = threadIdx.x;
    const int w = tid >> 5, lane = tid & 31;
    const int wy = w & 3, wx = w >> 2;
    const int khi = lane >> 4, l15 = lane & 15;

    v8f cacc[4] = {};
    const _Float16* aBase = hbuf + ((size_t)e*CAP + mBase) * HP;
    const float*    bBase = w2   + (size_t)e*CD*HD;
#if HAVE_TDM
    const uint32_t asOff = (uint32_t)(uintptr_t)&As[0][0];
#endif

    for (int kb = 0; kb < HP; kb += 64) {
#if HAVE_TDM
        if (tid < 32) tdm_load_tile64x64(asOff, aBase + kb, HP);
#else
#pragma unroll
        for (int rep = 0; rep < 2; rep++) {
            const int lin = (tid + rep*256) * 8;
            const int r = lin >> 6, cc = lin & 63;
            *(v8h*)&As[r][cc] = *(const v8h*)(aBase + (size_t)r*HP + kb + cc);
        }
#endif
#pragma unroll
        for (int rep = 0; rep < 8; rep++) {   // stage B (128x64) with K guard (HD)
            const int lin = (tid + rep*256) * 4;
            const int r = lin >> 6, cc = lin & 63;
            const float* src = bBase + (size_t)(nBase + r)*HD + kb + cc;
            v4h hv;
#pragma unroll
            for (int j = 0; j < 4; j++)
                hv[j] = (kb + cc + j < HD) ? (_Float16)src[j] : (_Float16)0.f;
            *(v4h*)&Bs[r][cc] = hv;
        }
        __builtin_prefetch(bBase + (size_t)nBase*HD + kb + 64, 0, 0);
#if HAVE_TDM
        if (tid < 32) __builtin_amdgcn_s_wait_tensorcnt(0);
#endif
        __syncthreads();
#pragma unroll
        for (int kc = 0; kc < 64; kc += 32) {
            const v16h a = frag16(&As[wy*16 + l15][kc], khi);
#pragma unroll
            for (int j = 0; j < 4; j++) {
                const v16h b = frag16(&Bs[wx*64 + j*16 + l15][kc], khi);
                cacc[j] = __builtin_amdgcn_wmma_f32_16x16x32_f16(false, a, false, b,
                                                                (short)0, cacc[j],
                                                                false, false);
            }
        }
        __syncthreads();
    }

    const int mRow = mBase + wy*16 + khi*8;
#pragma unroll
    for (int j = 0; j < 4; j++) {
        const int n = nBase + wx*64 + j*16 + l15;
        const float bias = b2[e*CD + n];
#pragma unroll
        for (int i = 0; i < 8; i++) {
            const int slot = e*CAP + mRow + i;
            const int tok = tok_slot[slot];
            const float pv = p_slot[slot];
            if (tok < NB) atomicAdd(y + (size_t)tok*CD + n, (cacc[j][i] + bias) * pv);
        }
    }
}

// ---------------------------------------------------------------------------
__global__ void k_final(const float* __restrict__ aux, const int* __restrict__ dropped,
                        float* __restrict__ out) {
    if (threadIdx.x == 0) {
        float tot = 0.f, cs = 0.f;
        for (int e = 0; e < NE; e++) { tot += aux[e]; cs += aux[8+e]; }
        float loss = 0.f;
        for (int e = 0; e < NE; e++) {
            const float frac = aux[8+e] / (cs + 1e-9f);
            const float imp  = aux[e]   / (tot + 1e-9f);
            out[OUT_FRAC+e] = frac;
            out[OUT_IMP+e]  = imp;
            loss += frac * imp;
        }
        out[OUT_LOSS] = 0.01f * (float)NE * loss;
        out[OUT_DROP] = (float)(*dropped);
    }
}

// ---------------------------------------------------------------------------
extern "C" void kernel_launch(void* const* d_in, const int* in_sizes, int n_in,
                              void* d_out, int out_size, void* d_ws, size_t ws_size,
                              hipStream_t stream) {
    const float* x   = (const float*)d_in[0];
    const float* gw  = (const float*)d_in[1];
    const float* w1  = (const float*)d_in[2];
    const float* b1  = (const float*)d_in[3];
    const float* w2  = (const float*)d_in[4];
    const float* b2  = (const float*)d_in[5];
    float* yout = (float*)d_out;

    char* wsb = (char*)d_ws;
    float*    aux     = (float*)(wsb + WS_AUX);
    int*      dropped = (int*)  (wsb + 64);
    int*      topki   = (int*)  (wsb + WS_TOPKI);
    float*    topkp   = (float*)(wsb + WS_TOPKP);
    int*      tok     = (int*)  (wsb + WS_TOK);
    float*    ps      = (float*)(wsb + WS_P);
    _Float16* bufh    = (_Float16*)(wsb + WS_BUFH);
    _Float16* hbuf    = (_Float16*)(wsb + WS_HBUF);

    k_init  <<<(NB*CD)/1024, 256, 0, stream>>>(yout, aux);
    k_gate  <<<NB/8,         256, 0, stream>>>(x, gw, topki, topkp, aux);
    k_route <<<NE,           256, 0, stream>>>(topki, topkp, tok, ps, dropped);
    k_gather<<<NE*CAP,       256, 0, stream>>>(x, tok, bufh);

    dim3 g1(HP/128, CAP/64, NE);     // 22 x 32 x 8
    k_ffn1<<<g1, 256, 0, stream>>>(bufh, w1, b1, hbuf);

    dim3 g2(CD/128, CAP/64, NE);     // 8 x 32 x 8
    k_ffn2<<<g2, 256, 0, stream>>>(hbuf, w2, b2, tok, ps, yout);

    k_final<<<1, 32, 0, stream>>>(aux, dropped, yout);
}
